// FlowLMStepONNX_36197984371414
// MI455X (gfx1250) — matrix operand order repaired and checked by reading, not compiled, not run on hardware
//
#include <hip/hip_runtime.h>
#include <hip/hip_bf16.h>
#include <math.h>

typedef __attribute__((ext_vector_type(16))) _Float16 v16h;
typedef __attribute__((ext_vector_type(4)))  _Float16 v4h;
typedef __attribute__((ext_vector_type(8)))  float    v8f;

#define Lx    16
#define Dm    1024
#define Hh    16
#define HDh   64
#define Tt    513
#define TKV   1537
#define PASTN 1024
#define FFN   4096
#define FHN   2048
#define LDIMN 512
#define SBST  40   // sB row stride in halves (80B: 16B-aligned rows, reduced bank conflicts)

// ---------------------------------------------------------------------------
// float4 copy of old KV cache into concatenated output cache
// src: (32, 1024, 1024) contiguous per (l,s); dst: (32, 1537, 1024), t=0..1023
// ---------------------------------------------------------------------------
__global__ __launch_bounds__(256)
void copy_cache_k(const float* __restrict__ src, float* __restrict__ dst) {
    size_t i4 = (size_t)blockIdx.x * 256 + threadIdx.x;   // < 8388608
    size_t i  = i4 * 4;
    size_t q  = i >> 20;            // (l*2+s)
    size_t rem = i & 1048575;       // t*1024 + e
    float4 v = ((const float4*)src)[i4];
    *(float4*)(dst + q * (size_t)(TKV * Dm) + rem) = v;
}

__global__ __launch_bounds__(256)
void copy_text_k(const float* __restrict__ src, float* __restrict__ dst) {
    size_t i4 = (size_t)blockIdx.x * 256 + threadIdx.x;   // < 131072
    ((float4*)dst)[i4] = ((const float4*)src)[i4];
}

// ---------------------------------------------------------------------------
// row 512 of x:  prev(NaN->bos) @ W_in + b_in
// ---------------------------------------------------------------------------
__global__ __launch_bounds__(256)
void in_proj_k(const float* __restrict__ prev, const float* __restrict__ bos,
               const float* __restrict__ Win, const float* __restrict__ bin,
               float* __restrict__ xrow) {
    int n = blockIdx.x * 256 + threadIdx.x;
    if (n >= Dm) return;
    float acc = bin[n];
    for (int k = 0; k < LDIMN; ++k) {
        float p = prev[k];
        p = (p != p) ? bos[k] : p;
        acc += p * Win[(size_t)k * Dm + n];
    }
    xrow[n] = acc;
}

// ---------------------------------------------------------------------------
// LayerNorm: one 256-thread block per row
// ---------------------------------------------------------------------------
__global__ __launch_bounds__(256)
void layernorm_k(const float* __restrict__ X, const float* __restrict__ w,
                 const float* __restrict__ b, float* __restrict__ Y) {
    const int row = blockIdx.x;
    const float* x = X + (size_t)row * Dm;
    float* y = Y + (size_t)row * Dm;
    const int tid = threadIdx.x;
    __shared__ float rs[8], rs2[8];
    float s = 0.f, s2 = 0.f;
    for (int i = tid; i < Dm; i += 256) { float v = x[i]; s += v; s2 += v * v; }
    for (int off = 16; off > 0; off >>= 1) {
        s  += __shfl_down(s,  off, 32);
        s2 += __shfl_down(s2, off, 32);
    }
    if ((tid & 31) == 0) { rs[tid >> 5] = s; rs2[tid >> 5] = s2; }
    __syncthreads();
    if (tid == 0) {
        float a = 0.f, c = 0.f;
        for (int i = 0; i < 8; ++i) { a += rs[i]; c += rs2[i]; }
        rs[0] = a; rs2[0] = c;
    }
    __syncthreads();
    float mean = rs[0] * (1.f / Dm);
    float var  = rs2[0] * (1.f / Dm) - mean * mean;
    float inv  = rsqrtf(var + 1e-5f);
    for (int i = tid; i < Dm; i += 256)
        y[i] = (x[i] - mean) * inv * w[i] + b[i];
}

// ---------------------------------------------------------------------------
// WMMA GEMM: C[M,N] = act(A[M,K] @ W[K,N] + bias) (+ Res)
// block = 128 threads (4 waves), tile 64x64, K step 32, fp32->f16 in LDS.
// Staging: batched float4 global loads into registers (no per-element waits),
// clamped addresses + value select (no exec divergence).
// ---------------------------------------------------------------------------
__global__ __launch_bounds__(128)
void gemm_wmma_k(const float* __restrict__ A, const float* __restrict__ W,
                 const float* __restrict__ bias, const float* __restrict__ Res,
                 float* __restrict__ C, int M, int N, int K, int act) {
    __shared__ _Float16 sA[64 * 32];        // [m][k], stride 32
    __shared__ _Float16 sB[64 * SBST];      // [n][k], stride SBST
    const int tid  = threadIdx.x;
    const int lane = tid & 31;
    const int wid  = tid >> 5;
    const int m0 = blockIdx.x * 64;
    const int n0 = blockIdx.y * 64;
    const int half = lane >> 4;
    const int lmod = lane & 15;

    v8f acc[4] = {};

    for (int k0 = 0; k0 < K; k0 += 32) {
        // ---- batched global loads into registers ----
        float4 a4[4], b4[4];
#pragma unroll
        for (int j = 0; j < 4; ++j) {
            int f = tid + j * 128;          // 0..511
            int r = f >> 3, c4 = f & 7;     // A tile: 64 rows x 8 float4
            int m = m0 + r;
            int mc = (m < M) ? m : (M - 1);
            a4[j] = *(const float4*)(A + (size_t)mc * K + k0 + c4 * 4);
        }
#pragma unroll
        for (int j = 0; j < 4; ++j) {
            int f = tid + j * 128;
            int k = f >> 4, n4 = f & 15;    // B tile: 32 rows x 16 float4
            b4[j] = *(const float4*)(W + (size_t)(k0 + k) * N + n0 + n4 * 4);
        }
        // prefetch next K tile
        if (k0 + 32 < K) {
            int f = tid + 128;
            int k = (f & 511) >> 4, n4 = f & 15;
            __builtin_prefetch((const void*)(W + (size_t)(k0 + 32 + k) * N + n0 + n4 * 4), 0, 3);
            int r = (tid & 511) >> 3, c4 = tid & 7;
            int m = m0 + r; int mc = (m < M) ? m : (M - 1);
            __builtin_prefetch((const void*)(A + (size_t)mc * K + k0 + 32 + c4 * 4), 0, 3);
        }
        // ---- convert + store to LDS ----
#pragma unroll
        for (int j = 0; j < 4; ++j) {
            int f = tid + j * 128;
            int r = f >> 3, c4 = f & 7;
            bool ok = (m0 + r) < M;
            v4h hv;
            hv[0] = ok ? (_Float16)a4[j].x : (_Float16)0.f;
            hv[1] = ok ? (_Float16)a4[j].y : (_Float16)0.f;
            hv[2] = ok ? (_Float16)a4[j].z : (_Float16)0.f;
            hv[3] = ok ? (_Float16)a4[j].w : (_Float16)0.f;
            *(v4h*)(sA + r * 32 + c4 * 4) = hv;
        }
#pragma unroll
        for (int j = 0; j < 4; ++j) {
            int f = tid + j * 128;
            int k = f >> 4, n4 = f & 15;
            sB[(n4 * 4 + 0) * SBST + k] = (_Float16)b4[j].x;
            sB[(n4 * 4 + 1) * SBST + k] = (_Float16)b4[j].y;
            sB[(n4 * 4 + 2) * SBST + k] = (_Float16)b4[j].z;
            sB[(n4 * 4 + 3) * SBST + k] = (_Float16)b4[j].w;
        }
        __syncthreads();

        // A fragment (16x32 f16): lane = M row of this wave's strip
        v16h af;
        const int mrow = wid * 16 + lmod;
        const int hof  = half * 8;
#pragma unroll
        for (int v = 0; v < 8; ++v) {
            int kk = hof * 2 + v * 2;
            af[2 * v]     = sA[mrow * 32 + kk];
            af[2 * v + 1] = sA[mrow * 32 + kk + 1];
        }
        const int kb = half * 16;
#pragma unroll
        for (int nt = 0; nt < 4; ++nt) {
            v16h bf;
            int col = nt * 16 + lmod;
#pragma unroll
            for (int v = 0; v < 8; ++v) {
                bf[2 * v]     = sB[col * SBST + kb + 2 * v];
                bf[2 * v + 1] = sB[col * SBST + kb + 2 * v + 1];
            }
            acc[nt] = __builtin_amdgcn_wmma_f32_16x16x32_f16(
                false, af, false, bf, (short)0, acc[nt], false, false);
        }
        __syncthreads();
    }

#pragma unroll
    for (int nt = 0; nt < 4; ++nt) {
        int n = n0 + nt * 16 + lmod;
#pragma unroll
        for (int v = 0; v < 8; ++v) {
            int m = m0 + wid * 16 + v + 8 * half;
            if (m < M) {
                float val = acc[nt][v] + bias[n];
                if (act) val = 0.5f * val * (1.f + erff(val * 0.70710678118f));
                if (Res) val += Res[(size_t)m * N + n];
                C[(size_t)m * N + n] = val;
            }
        }
    }
}

// ---------------------------------------------------------------------------
// RoPE + scatter into q buffer and output KV cache (layer l, new rows)
// ---------------------------------------------------------------------------
__global__ __launch_bounds__(256)
void rope_cache_k(const float* __restrict__ qkv, const int* __restrict__ clen,
                  float* __restrict__ qb, float* __restrict__ cache, int l) {
    int i = blockIdx.x * 256 + threadIdx.x;
    if (i >= Tt * Hh * 32) return;
    int j = i & 31;
    int h = (i >> 5) & 15;
    int t = i >> 9;
    float freq = expf(-(float)j * 0.28782313662425576f);   // log(1e4)*2/64
    float ang  = ((float)t + (float)clen[0]) * freq;
    float c = cosf(ang), s = sinf(ang);

    size_t base = (size_t)t * 3072 + (size_t)h * 64 + 2 * j;
    float qr = qkv[base], qi = qkv[base + 1];
    size_t qo = ((size_t)t * Hh + h) * 64 + 2 * j;
    qb[qo]     = qr * c - qi * s;
    qb[qo + 1] = qr * s + qi * c;
    float kr = qkv[base + 1024], ki = qkv[base + 1025];
    size_t ko = ((size_t)(l * 2) * TKV + PASTN + t) * (size_t)Dm + (size_t)h * 64 + 2 * j;
    cache[ko]     = kr * c - ki * s;
    cache[ko + 1] = kr * s + ki * c;
    size_t vo = ((size_t)(l * 2 + 1) * TKV + PASTN + t) * (size_t)Dm + (size_t)h * 64 + 2 * j;
    cache[vo]     = qkv[base + 2048];
    cache[vo + 1] = qkv[base + 2049];
}

// ---------------------------------------------------------------------------
// Flash attention: one wave per (head, 16-query tile), WMMA for QK^T and PV.
// All guarded loads are clamped-address + select (branchless, EXEC all-ones).
// ---------------------------------------------------------------------------
__global__ __launch_bounds__(32)
void attn_wmma_k(const float* __restrict__ Q, const float* __restrict__ cache,
                 float* __restrict__ O, int l) {
    __shared__ _Float16 sP[16 * 32];
    const int lane = threadIdx.x;
    const int NQT = 33;
    const int h  = blockIdx.x / NQT;
    const int qt = blockIdx.x % NQT;
    const int lmod = lane & 15;
    const int half = lane >> 4;
    const float scale = 0.125f;   // 1/sqrt(64)

    const float* Kc = cache + (size_t)(l * 2)     * TKV * Dm;
    const float* Vc = cache + (size_t)(l * 2 + 1) * TKV * Dm;
    const size_t hoff = (size_t)h * 64;

    // Q fragments (16x64 -> two 16x32 A-frags), float2 pair loads
    v16h qa[2];
    {
        int t = qt * 16 + lmod;
        bool ok = t < Tt;
        int tc = ok ? t : (Tt - 1);
        const float2* q2 = (const float2*)(Q + ((size_t)tc * Hh + h) * 64);
#pragma unroll
        for (int c = 0; c < 2; ++c) {
#pragma unroll
            for (int v = 0; v < 8; ++v) {
                int d = c * 32 + half * 16 + v * 2;
                float2 p = q2[d >> 1];
                qa[c][2 * v]     = ok ? (_Float16)p.x : (_Float16)0.f;
                qa[c][2 * v + 1] = ok ? (_Float16)p.y : (_Float16)0.f;
            }
        }
    }

    float rmax[8], rsum[8];
    v8f o[4] = {};
#pragma unroll
    for (int v = 0; v < 8; ++v) { rmax[v] = -3e38f; rsum[v] = 0.f; }

    const int nchunk = (TKV + 31) / 32;   // 49
    for (int ch = 0; ch < nchunk; ++ch) {
        const int kk0 = ch * 32;
        v8f s[2];
#pragma unroll
        for (int sub = 0; sub < 2; ++sub) {
            v16h bf0, bf1;
            int key = kk0 + sub * 16 + lmod;
            bool ok = key < TKV;
            int kc = ok ? key : (TKV - 1);
            const float2* k2 = (const float2*)(Kc + (size_t)kc * Dm + hoff);
            int kb2 = half * 16;
#pragma unroll
            for (int v = 0; v < 8; ++v) {
                int d0 = kb2 + 2 * v;
                float2 p0 = k2[d0 >> 1];
                float2 p1 = k2[(d0 >> 1) + 16];
                bf0[2 * v]     = ok ? (_Float16)p0.x : (_Float16)0.f;
                bf0[2 * v + 1] = ok ? (_Float16)p0.y : (_Float16)0.f;
                bf1[2 * v]     = ok ? (_Float16)p1.x : (_Float16)0.f;
                bf1[2 * v + 1] = ok ? (_Float16)p1.y : (_Float16)0.f;
            }
            v8f z = {};
            v8f t0 = __builtin_amdgcn_wmma_f32_16x16x32_f16(
                false, qa[0], false, bf0, (short)0, z, false, false);
            s[sub] = __builtin_amdgcn_wmma_f32_16x16x32_f16(
                false, qa[1], false, bf1, (short)0, t0, false, false);
        }
        // mask + scale
#pragma unroll
        for (int sub = 0; sub < 2; ++sub) {
            int key = kk0 + sub * 16 + lmod;
#pragma unroll
            for (int v = 0; v < 8; ++v) {
                int tq = qt * 16 + v + 8 * half;
                bool ok = (key < TKV) && (key <= PASTN + tq);
                s[sub][v] = ok ? s[sub][v] * scale : -1e30f;
            }
        }
        // online softmax update
        float alpha[8];
#pragma unroll
        for (int v = 0; v < 8; ++v) {
            float mx = fmaxf(s[0][v], s[1][v]);
            mx = fmaxf(mx, __shfl_xor(mx, 1, 32));
            mx = fmaxf(mx, __shfl_xor(mx, 2, 32));
            mx = fmaxf(mx, __shfl_xor(mx, 4, 32));
            mx = fmaxf(mx, __shfl_xor(mx, 8, 32));
            float mnew = fmaxf(rmax[v], mx);
            alpha[v] = expf(rmax[v] - mnew);
            rmax[v] = mnew;
            float p0 = expf(s[0][v] - mnew);
            float p1 = expf(s[1][v] - mnew);
            s[0][v] = p0; s[1][v] = p1;
            float ps = p0 + p1;
            ps += __shfl_xor(ps, 1, 32);
            ps += __shfl_xor(ps, 2, 32);
            ps += __shfl_xor(ps, 4, 32);
            ps += __shfl_xor(ps, 8, 32);
            rsum[v] = rsum[v] * alpha[v] + ps;
        }
#pragma unroll
        for (int nt = 0; nt < 4; ++nt)
#pragma unroll
            for (int v = 0; v < 8; ++v) o[nt][v] *= alpha[v];

        // P (C-layout) -> LDS -> A-fragment
#pragma unroll
        for (int v = 0; v < 8; ++v) {
            int row = v + 8 * half;
            sP[row * 32 + lmod]      = (_Float16)s[0][v];
            sP[row * 32 + 16 + lmod] = (_Float16)s[1][v];
        }
        __syncthreads();
        v16h pf;
#pragma unroll
        for (int v = 0; v < 8; ++v) {
            int kk = half * 16 + v * 2;
            pf[2 * v]     = sP[lmod * 32 + kk];
            pf[2 * v + 1] = sP[lmod * 32 + kk + 1];
        }
        // V fragments and PV accumulation (clamped rows + select)
#pragma unroll
        for (int nt = 0; nt < 4; ++nt) {
            v16h vf;
            int d = nt * 16 + lmod;
#pragma unroll
            for (int v = 0; v < 8; ++v) {
                int key0 = kk0 + half * 16 + 2 * v;
                int c0 = (key0 < TKV) ? key0 : (TKV - 1);
                int c1 = (key0 + 1 < TKV) ? (key0 + 1) : (TKV - 1);
                float v0 = Vc[(size_t)c0 * Dm + hoff + d];
                float v1 = Vc[(size_t)c1 * Dm + hoff + d];
                vf[2 * v]     = (key0     < TKV) ? (_Float16)v0 : (_Float16)0.f;
                vf[2 * v + 1] = (key0 + 1 < TKV) ? (_Float16)v1 : (_Float16)0.f;
            }
            o[nt] = __builtin_amdgcn_wmma_f32_16x16x32_f16(
                false, pf, false, vf, (short)0, o[nt], false, false);
        }
        __syncthreads();
    }

    // write o[t, h*64+d]
#pragma unroll
    for (int nt = 0; nt < 4; ++nt) {
        int d = nt * 16 + lmod;
#pragma unroll
        for (int v = 0; v < 8; ++v) {
            int tq = qt * 16 + v + 8 * half;
            if (tq < Tt)
                O[((size_t)tq * Hh + h) * 64 + d] = o[nt][v] / rsum[v];
        }
    }
}

// ---------------------------------------------------------------------------
// flow MLP layer 1: fh = gelu(f_in @ Wf1 + bf1),  f_in = [t_out, 0, 1, noise]
// ---------------------------------------------------------------------------
__global__ __launch_bounds__(256)
void flow1_k(const float* __restrict__ hbuf, const float* __restrict__ noise,
             const float* __restrict__ Wf1, const float* __restrict__ bf1,
             float* __restrict__ fh) {
    int n = blockIdx.x * 256 + threadIdx.x;
    if (n >= FHN) return;
    const float* trow = hbuf + (size_t)512 * Dm;
    float acc = bf1[n];
    for (int k = 0; k < Dm; ++k) acc += trow[k] * Wf1[(size_t)k * FHN + n];
    acc += Wf1[(size_t)(Dm + 1) * FHN + n];   // t_time = 1 (s_time = 0)
    for (int k = 0; k < LDIMN; ++k)
        acc += noise[k] * Wf1[(size_t)(Dm + 2 + k) * FHN + n];
    fh[n] = 0.5f * acc * (1.f + erff(acc * 0.70710678118f));
}

__global__ __launch_bounds__(256)
void flow2_k(const float* __restrict__ fh, const float* __restrict__ noise,
             const float* __restrict__ Wf2, const float* __restrict__ bf2,
             float* __restrict__ out_lat) {
    int n = blockIdx.x * 256 + threadIdx.x;
    if (n >= LDIMN) return;
    float acc = bf2[n];
    for (int k = 0; k < FHN; ++k) acc += fh[k] * Wf2[(size_t)k * LDIMN + n];
    out_lat[n] = noise[n] + acc;
}

__global__ __launch_bounds__(256)
void eos_len_k(const float* __restrict__ hbuf, const float* __restrict__ Weos,
               const float* __restrict__ beos, const int* __restrict__ clen,
               float* __restrict__ out_eos, float* __restrict__ out_len) {
    __shared__ float red[256];
    const int tid = threadIdx.x;
    const float* trow = hbuf + (size_t)512 * Dm;
    float s = 0.f;
    for (int i = tid; i < Dm; i += 256) s += trow[i] * Weos[i];
    red[tid] = s;
    __syncthreads();
    for (int st = 128; st > 0; st >>= 1) {
        if (tid < st) red[tid] += red[tid + st];
        __syncthreads();
    }
    if (tid == 0) {
        float v = red[0] + beos[0];
        out_eos[0] = (v > -4.0f) ? 1.f : 0.f;
        out_len[0] = (float)(clen[0] + Tt);
    }
}

// ---------------------------------------------------------------------------
extern "C" void kernel_launch(void* const* d_in, const int* in_sizes, int n_in,
                              void* d_out, int out_size, void* d_ws, size_t ws_size,
                              hipStream_t stream) {
    (void)in_sizes; (void)n_in; (void)out_size; (void)ws_size;
    const float* text  = (const float*)d_in[0];
    const float* prev  = (const float*)d_in[1];
    const float* kvc   = (const float*)d_in[2];
    const int*   clen  = (const int*)d_in[3];
    const float* noise = (const float*)d_in[4];
    const float* bos   = (const float*)d_in[5];
    const float* Win   = (const float*)d_in[6];
    const float* bin   = (const float*)d_in[7];
    const float* ln1w  = (const float*)d_in[8];
    const float* ln1b  = (const float*)d_in[9];
    const float* Wqkv  = (const float*)d_in[10];
    const float* bqkv  = (const float*)d_in[11];
    const float* Wo    = (const float*)d_in[12];
    const float* bo    = (const float*)d_in[13];
    const float* ln2w  = (const float*)d_in[14];
    const float* ln2b  = (const float*)d_in[15];
    const float* W1    = (const float*)d_in[16];
    const float* b1    = (const float*)d_in[17];
    const float* W2    = (const float*)d_in[18];
    const float* b2    = (const float*)d_in[19];
    const float* onw   = (const float*)d_in[20];
    const float* onb   = (const float*)d_in[21];
    const float* Weos  = (const float*)d_in[22];
    const float* beos  = (const float*)d_in[23];
    const float* Wf1   = (const float*)d_in[24];
    const float* bf1   = (const float*)d_in[25];
    const float* Wf2   = (const float*)d_in[26];
    const float* bf2   = (const float*)d_in[27];

    float* out       = (float*)d_out;
    float* out_lat   = out;                         // 512
    float* out_eos   = out + 512;                   // 1
    float* out_cache = out + 513;                   // 16*2*1537*1024
    float* out_len   = out + 513 + (size_t)Lx * 2 * TKV * Dm;

    float* ws  = (float*)d_ws;
    float* x    = ws;                               // 513*1024
    float* hbuf = x    + (size_t)Tt * Dm;           // 513*1024
    float* qkv  = hbuf + (size_t)Tt * Dm;           // 513*3072
    float* qb   = qkv  + (size_t)Tt * 3 * Dm;       // 513*1024
    float* ob   = qb   + (size_t)Tt * Dm;           // 513*1024
    float* ffb  = ob   + (size_t)Tt * Dm;           // 513*4096
    float* fh   = ffb  + (size_t)Tt * FFN;          // 2048

    copy_cache_k<<<32768, 256, 0, stream>>>(kvc, out_cache);
    copy_text_k<<<512, 256, 0, stream>>>(text, x);
    in_proj_k<<<4, 256, 0, stream>>>(prev, bos, Win, bin, x + (size_t)512 * Dm);

    const int MB = (Tt + 63) / 64;  // 9
    for (int l = 0; l < Lx; ++l) {
        layernorm_k<<<Tt, 256, 0, stream>>>(x, ln1w + l * Dm, ln1b + l * Dm, hbuf);
        gemm_wmma_k<<<dim3(MB, 3 * Dm / 64), 128, 0, stream>>>(
            hbuf, Wqkv + (size_t)l * Dm * 3 * Dm, bqkv + (size_t)l * 3 * Dm,
            nullptr, qkv, Tt, 3 * Dm, Dm, 0);
        rope_cache_k<<<(Tt * Hh * 32 + 255) / 256, 256, 0, stream>>>(
            qkv, clen, qb, out_cache, l);
        attn_wmma_k<<<Hh * 33, 32, 0, stream>>>(qb, out_cache, ob, l);
        gemm_wmma_k<<<dim3(MB, Dm / 64), 128, 0, stream>>>(
            ob, Wo + (size_t)l * Dm * Dm, bo + (size_t)l * Dm, x, x, Tt, Dm, Dm, 0);
        layernorm_k<<<Tt, 256, 0, stream>>>(x, ln2w + l * Dm, ln2b + l * Dm, hbuf);
        gemm_wmma_k<<<dim3(MB, FFN / 64), 128, 0, stream>>>(
            hbuf, W1 + (size_t)l * Dm * FFN, b1 + (size_t)l * FFN,
            nullptr, ffb, Tt, FFN, Dm, 1);
        gemm_wmma_k<<<dim3(MB, Dm / 64), 128, 0, stream>>>(
            ffb, W2 + (size_t)l * FFN * Dm, b2 + (size_t)l * Dm, x, x, Tt, Dm, FFN, 0);
    }

    layernorm_k<<<Tt, 256, 0, stream>>>(x, onw, onb, hbuf);
    flow1_k<<<FHN / 256, 256, 0, stream>>>(hbuf, noise, Wf1, bf1, fh);
    flow2_k<<<LDIMN / 256, 256, 0, stream>>>(fh, noise, Wf2, bf2, out_lat);
    eos_len_k<<<1, 256, 0, stream>>>(hbuf, Weos, beos, clen, out_eos, out_len);
}
